// NeighConsensus_47974784696830
// MI455X (gfx1250) — matrix operand order repaired and checked by reading, not compiled, not run on hardware
//
#include <hip/hip_runtime.h>

typedef __attribute__((ext_vector_type(16))) __bf16 v16bf;
typedef __attribute__((ext_vector_type(8)))  __bf16 v8bf;
typedef __attribute__((ext_vector_type(8)))  float  v8f;

#define TPAD 18          // t' extent: 16-tile + 2 halo
#define DBLK 8           // d values per workgroup
#define DPAD (DBLK + 2)  // d halo
#define GPAD 96          // (ci,kh,kw) groups padded to 3 chunks of 32
#define MAXFRAG 27       // 3 kt * 3 kd * 3 chunks

// Implicit-GEMM 4D conv layer: out = relu(conv4d(in, w) + bias)
// in : [B, CIN, 32,32,32,32] f32 (PERM=1: read (H,W)<->(D,T)-permuted)
// w  : [3, COUT, CIN, 3,3,3] f32
// out: [B, COUT, 32,32,32,32] f32 (MODE=1: accumulate at transposed index)
template <int CIN, int COUT, int PERM, int MODE>
__global__ __launch_bounds__(256)
void conv4d_wmma(const float* __restrict__ in, const float* __restrict__ wgt,
                 const float* __restrict__ bias, float* __restrict__ out)
{
    constexpr int G      = CIN * 9;
    constexpr int NCHUNK = (G + 31) / 32;       // 1 (CIN=1) or 3 (CIN=10)
    constexpr int NFRAG  = 9 * NCHUNK;

    __shared__ __align__(16) __bf16 lin[TPAD * DPAD * GPAD];   // 34,560 B
    // B fragments, per-lane contiguous: [frag][lane(32)][e(16)] bf16
    __shared__ __align__(16) __bf16 lw[MAXFRAG * 32 * 16];     // 27,648 B

    const int tid = threadIdx.x;
    unsigned bx = blockIdx.x;
    const int thalf = bx & 1;  bx >>= 1;
    const int dblk  = bx & 3;  bx >>= 2;
    const int w     = bx & 31; bx >>= 5;
    const int h     = bx & 31; bx >>= 5;
    const int b     = bx;            // 0..3
    const int d0 = dblk * DBLK;
    const int t0 = thalf * 16;

    // ---- stage input halo slab into LDS (bf16), layout [t'][dd][g] ----
    for (int idx = tid; idx < TPAD * DPAD * GPAD; idx += 256) {
        int g    = idx % GPAD;
        int rest = idx / GPAD;
        int dd   = rest % DPAD;
        int tp   = rest / DPAD;
        float v = 0.0f;
        if (g < G) {
            int ci = g / 9;
            int r9 = g - ci * 9;
            int kh = r9 / 3;
            int kw = r9 - kh * 3;
            int hh = h + kh - 1;
            int ww = w + kw - 1;
            int dI = d0 + dd - 1;
            int tI = t0 + tp - 1;
            if ((unsigned)hh < 32u && (unsigned)ww < 32u &&
                (unsigned)dI < 32u && (unsigned)tI < 32u) {
                long base = ((long)((long)b * CIN + ci)) << 20;   // * 32^4
                long off = PERM
                    ? (((((long)dI * 32 + tI) * 32 + hh) * 32) + ww)   // x[b,ci,d,t,h,w]
                    : (((((long)hh * 32 + ww) * 32 + dI) * 32) + tI);  // x[b,ci,h,w,d,t]
                v = in[base + off];
            }
        }
        lin[idx] = (__bf16)v;
    }

    // ---- stage weights: lw[frag][lane][e] = B[K = (e>=8)*16 + (lane>=16)*8 + (e&7)][c = lane&15] ----
    for (int idx = tid; idx < NFRAG * 512; idx += 256) {
        int e    = idx & 15;
        int ln   = (idx >> 4) & 31;
        int frag = idx >> 9;
        int chunk = frag % NCHUNK;
        int rest  = frag / NCHUNK;
        int kd = rest % 3;
        int kt = rest / 3;
        int k  = ((e >> 3) << 4) + ((ln >> 4) << 3) + (e & 7);      // K within chunk: 0..31
        int c  = ln & 15;                   // cout column
        int gk = chunk * 32 + k;
        float v = 0.0f;
        if (gk < G && c < COUT) {
            int ci = gk / 9;
            int r9 = gk - ci * 9;
            int kh = r9 / 3;
            int kw = r9 - kh * 3;
            // w[kh][c][ci][kw][kd][kt]
            v = wgt[((((long)kh * COUT + c) * CIN + ci) * 27) + (long)kw * 9 + kd * 3 + kt];
        }
        lw[idx] = (__bf16)v;
    }

    __syncthreads();

    // ---- per-wave implicit GEMM: 16 t-positions (M) x 16 couts (N) ----
    const int wave   = tid >> 5;        // 0..7 -> d = d0 + wave
    const int lane   = tid & 31;
    const int hiHalf = (lane >> 4) & 1; // 0: lanes 0-15, 1: lanes 16-31
    const int col    = lane & 15;       // N (cout) / M-row for A

    v8f acc = {0.f, 0.f, 0.f, 0.f, 0.f, 0.f, 0.f, 0.f};

    const __bf16* bbase = lw + lane * 16;
    #pragma unroll
    for (int kt = 0; kt < 3; ++kt) {
        #pragma unroll
        for (int kd = 0; kd < 3; ++kd) {
            const int tp = col + kt;        // t' = (t - t0) + kt, m = col
            const int dd = wave + kd;       // (d - d0) + kd
            const __bf16* arow = lin + ((tp * DPAD) + dd) * GPAD + hiHalf * 8;
            #pragma unroll
            for (int chunk = 0; chunk < NCHUNK; ++chunk) {
                const int frag = (kt * 3 + kd) * NCHUNK + chunk;

                // B fragment: one contiguous 32B run per lane -> 2x ds_load_b128
                const __bf16* bp = bbase + frag * 512;
                v8bf b1 = *(const v8bf*)(bp);
                v8bf b2 = *(const v8bf*)(bp + 8);
                v16bf bf = __builtin_shufflevector(b1, b2, 0, 1, 2, 3, 4, 5, 6, 7,
                                                           8, 9, 10, 11, 12, 13, 14, 15);

                // A fragment: two 16B runs per lane -> 2x ds_load_b128
                const __bf16* ap = arow + chunk * 32;
                v8bf a1 = *(const v8bf*)(ap);        // K +0..7   (or +8..15)
                v8bf a2 = *(const v8bf*)(ap + 16);   // K +16..23 (or +24..31)
                v16bf af = __builtin_shufflevector(a1, a2, 0, 1, 2, 3, 4, 5, 6, 7,
                                                           8, 9, 10, 11, 12, 13, 14, 15);

                acc = __builtin_amdgcn_wmma_f32_16x16x32_bf16(
                          false, af, false, bf, (short)0, acc, false, false);
            }
        }
    }

    // ---- bias + ReLU + store (C/D: VGPR r -> M = r + 8*hiHalf, N = col) ----
    if (col < COUT) {
        const int d = d0 + wave;
        const float bv = bias[col];
        #pragma unroll
        for (int r = 0; r < 8; ++r) {
            int t = t0 + r + 8 * hiHalf;
            float v = acc[r] + bv;
            v = v > 0.f ? v : 0.f;
            if (MODE == 0) {
                long idx = ((((long)b * COUT + col) * 32 + h) * 32 + w);
                idx = (idx * 32 + d) * 32 + t;
                out[idx] = v;
            } else {
                // permuted pass, last layer: out[b,c,d,t,h,w] += v
                long idx = ((((long)b * COUT + col) * 32 + d) * 32 + t);
                idx = (idx * 32 + h) * 32 + w;
                out[idx] += v;
            }
        }
    }
}

// grid = b(4) * h(32) * w(32) * dblk(4) * thalf(2)
#define CONV_GRID dim3(4 * 32 * 32 * 4 * 2)

extern "C" void kernel_launch(void* const* d_in, const int* in_sizes, int n_in,
                              void* d_out, int out_size, void* d_ws, size_t ws_size,
                              hipStream_t stream)
{
    const float* x  = (const float*)d_in[0];
    const float* w0 = (const float*)d_in[1];
    const float* b0 = (const float*)d_in[2];
    const float* w1 = (const float*)d_in[3];
    const float* b1 = (const float*)d_in[4];
    const float* w2 = (const float*)d_in[5];
    const float* b2 = (const float*)d_in[6];
    float* out = (float*)d_out;

    const size_t NCH10 = (size_t)4 * 10 * 32 * 32 * 32 * 32; // 41,943,040 floats
    float* buf1 = (float*)d_ws;
    float* buf2 = buf1 + NCH10;

    // pass A: plain orientation; layer 2 overwrites d_out entirely
    conv4d_wmma<1, 10, 0, 0><<<CONV_GRID, 256, 0, stream>>>(x,    w0, b0, buf1);
    conv4d_wmma<10, 10, 0, 0><<<CONV_GRID, 256, 0, stream>>>(buf1, w1, b1, buf2);
    conv4d_wmma<10, 1, 0, 0><<<CONV_GRID, 256, 0, stream>>>(buf2, w2, b2, out);

    // pass B: (H,W)<->(D,T) permuted input; layer 2 accumulates transposed
    conv4d_wmma<1, 10, 1, 0><<<CONV_GRID, 256, 0, stream>>>(x,    w0, b0, buf1);
    conv4d_wmma<10, 10, 0, 0><<<CONV_GRID, 256, 0, stream>>>(buf1, w1, b1, buf2);
    conv4d_wmma<10, 1, 0, 1><<<CONV_GRID, 256, 0, stream>>>(buf2, w2, b2, out);
}
